// PSFM_43559558316832
// MI455X (gfx1250) — compile-verified, matrix operand order checked by reading, not run on hardware
//
#include <hip/hip_runtime.h>
#include <stdint.h>

typedef __attribute__((ext_vector_type(16))) __bf16 v16bf;
typedef __attribute__((ext_vector_type(8)))  float  v8f;
typedef unsigned short u16;

#define LDSP 72   // u16 pitch per LDS row: 64 + 8 pad -> 144B rows: 16B-aligned, bank-conflict free

__device__ __forceinline__ u16 f32_bf16(float f) {
    unsigned u = __float_as_uint(f);
    return (u16)((u + 0x7FFFu + ((u >> 16) & 1u)) >> 16);   // RNE
}

union Frag { uint4 q[2]; unsigned u[8]; v16bf v; };

// C[M,N] = A[M,K] * B[K,N] (+R), all-bf16 operands in WMMA-friendly layouts:
//   A: [M][K] bf16, K contiguous, pitch lda
//   B: stored transposed Bt: [N][K] bf16, K contiguous, pitch ldb
// Outputs: optional f32 C (pitch ldf, +optional f32 residual R same pitch),
//          optional bf16 C (pitch ldcb), optionally transposed ([N][M]).
// 128x128 macro-tile, 8 waves of 32 (wave tile 32x64), BK=64 per round.
__global__ __launch_bounds__(256)
void gemm_bf16(const u16* __restrict__ A, long long lda, long long sAb,
               const u16* __restrict__ B, long long ldb, long long sBb,
               float* __restrict__ Cf, const float* __restrict__ R,
               long long ldf, long long sCfb, long long sRb,
               u16* __restrict__ Cb, long long ldcb, long long sCbb, int transCb,
               int K)
{
    __shared__ __align__(16) u16 As[128 * LDSP];
    __shared__ __align__(16) u16 Bs[128 * LDSP];

    const int tid  = threadIdx.x;
    const int lane = tid & 31;
    const int wave = tid >> 5;
    const int wm   = wave & 3;    // 4 M-groups of 32 rows
    const int wn   = wave >> 2;   // 2 N-groups of 64 cols
    const int l    = lane & 15;
    const int hi   = lane >> 4;

    const int i0 = blockIdx.y * 128;
    const int j0 = blockIdx.x * 128;
    const int b  = blockIdx.z;

    const u16* Ag = A + (long long)b * sAb + (long long)i0 * lda;
    const u16* Bg = B + (long long)b * sBb + (long long)j0 * ldb;

    v8f zero = {0.f,0.f,0.f,0.f,0.f,0.f,0.f,0.f};
    v8f acc[2][4];
#pragma unroll
    for (int r = 0; r < 2; ++r)
#pragma unroll
        for (int c = 0; c < 4; ++c) acc[r][c] = zero;

    const char* AsB = (const char*)As;
    const char* BsB = (const char*)Bs;
    int arow[2], brow[4];
#pragma unroll
    for (int r = 0; r < 2; ++r) arow[r] = (wm * 32 + r * 16 + l) * (LDSP * 2);
#pragma unroll
    for (int c = 0; c < 4; ++c) brow[c] = (wn * 64 + c * 16 + l) * (LDSP * 2);
    const int hb = hi * 16;

    for (int k0 = 0; k0 < K; k0 += 64) {
        // cooperative staging: 2048 16-byte chunks (A rows + Bt rows), 8 per thread,
        // pure b128 global->LDS copies (producers already emitted bf16 in-layout)
#pragma unroll
        for (int s = 0; s < 8; ++s) {
            int it  = tid + s * 256;
            int isB = it >> 10;
            int r   = (it & 1023) >> 3;
            int ch  = it & 7;
            const u16* src = (isB ? Bg + (long long)r * ldb
                                  : Ag + (long long)r * lda) + k0 + ch * 8;
            uint4 d = *(const uint4*)src;
            u16* dst = (isB ? Bs : As) + r * LDSP + ch * 8;
            *(uint4*)dst = d;
        }
        __syncthreads();

#pragma unroll
        for (int ks = 0; ks < 2; ++ks) {
            const int kb = ks * 64 + hb;          // chunk0 byte offset within row
            Frag a[2], bt[4];
#pragma unroll
            for (int r = 0; r < 2; ++r) {         // 2x ds_load_b128 per fragment
                a[r].q[0] = *(const uint4*)(AsB + arow[r] + kb);
                a[r].q[1] = *(const uint4*)(AsB + arow[r] + kb + 32);
            }
#pragma unroll
            for (int c = 0; c < 4; ++c) {
                bt[c].q[0] = *(const uint4*)(BsB + brow[c] + kb);
                bt[c].q[1] = *(const uint4*)(BsB + brow[c] + kb + 32);
            }
#pragma unroll
            for (int r = 0; r < 2; ++r)
#pragma unroll
                for (int c = 0; c < 4; ++c)
                    acc[r][c] = __builtin_amdgcn_wmma_f32_16x16x32_bf16(
                        false, a[r].v, false, bt[c].v, (short)0, acc[r][c],
                        false, false);
        }
        __syncthreads();
    }

    const float* Rb  = R  ? (R  + (long long)b * sRb)  : nullptr;
    float*       Cfb = Cf ? (Cf + (long long)b * sCfb) : nullptr;
    u16*         Cbb = Cb ? (Cb + (long long)b * sCbb) : nullptr;

#pragma unroll
    for (int r = 0; r < 2; ++r) {
#pragma unroll
        for (int c = 0; c < 4; ++c) {
            int col = j0 + wn * 64 + c * 16 + l;
#pragma unroll
            for (int v = 0; v < 8; ++v) {
                int row = i0 + wm * 32 + r * 16 + v + hi * 8;
                float val = acc[r][c][v];
                long long fo = (long long)row * ldf + col;
                if (Rb)  val += Rb[fo];
                if (Cfb) Cfb[fo] = val;
                if (Cbb) {
                    long long bo = transCb ? ((long long)col * ldcb + row)
                                           : ((long long)row * ldcb + col);
                    Cbb[bo] = f32_bf16(val);
                }
            }
        }
    }
}

// elementwise f32 -> bf16 (weights)
__global__ __launch_bounds__(256)
void cvt_bf16(const float* __restrict__ in, u16* __restrict__ out, int n)
{
    int i = blockIdx.x * 256 + threadIdx.x;
    if (i < n) out[i] = f32_bf16(in[i]);
}

// tiled transpose + convert: in f32 [Cn][Np] per batch -> out bf16 [Np][Cn]
__global__ __launch_bounds__(256)
void transpose_cvt(const float* __restrict__ in, u16* __restrict__ out, int Cn, int Np)
{
    __shared__ u16 tile[32][33];
    int p0 = blockIdx.x * 32, c0 = blockIdx.y * 32, b = blockIdx.z;
    const float* inb = in + (long long)b * Cn * Np;
    u16* outb = out + (long long)b * Cn * Np;
    int tx = threadIdx.x & 31, ty = threadIdx.x >> 5;
#pragma unroll
    for (int i = 0; i < 4; ++i) {
        int r = ty + i * 8;
        tile[r][tx] = f32_bf16(inb[(long long)(c0 + r) * Np + p0 + tx]);
    }
    __syncthreads();
#pragma unroll
    for (int i = 0; i < 4; ++i) {
        int r = ty + i * 8;
        outb[(long long)(p0 + r) * Cn + c0 + tx] = tile[tx][r];
    }
}

// row softmax over 1024 cols; reads f32 S, writes bf16 probabilities only.
__global__ __launch_bounds__(256)
void softmax_bf16(const float* __restrict__ S, u16* __restrict__ Sb)
{
    __shared__ float red[256];
    const int tid = threadIdx.x;
    long long roff = ((long long)blockIdx.y * gridDim.x + blockIdx.x) * 1024;
    const float* row = S + roff;
    u16* rowb = Sb + roff;

    float v[4];
#pragma unroll
    for (int i = 0; i < 4; ++i) v[i] = row[tid + i * 256];
    float m = fmaxf(fmaxf(v[0], v[1]), fmaxf(v[2], v[3]));
    red[tid] = m; __syncthreads();
    for (int s = 128; s > 0; s >>= 1) {
        if (tid < s) red[tid] = fmaxf(red[tid], red[tid + s]);
        __syncthreads();
    }
    m = red[0]; __syncthreads();

    float sum = 0.f;
#pragma unroll
    for (int i = 0; i < 4; ++i) { v[i] = __expf(v[i] - m); sum += v[i]; }
    red[tid] = sum; __syncthreads();
    for (int s = 128; s > 0; s >>= 1) {
        if (tid < s) red[tid] += red[tid + s];
        __syncthreads();
    }
    float inv = 1.0f / red[0];
#pragma unroll
    for (int i = 0; i < 4; ++i) rowb[tid + i * 256] = f32_bf16(v[i] * inv);
}

extern "C" void kernel_launch(void* const* d_in, const int* in_sizes, int n_in,
                              void* d_out, int out_size, void* d_ws, size_t ws_size,
                              hipStream_t stream)
{
    (void)in_sizes; (void)n_in; (void)out_size; (void)ws_size;
    const float* rgb   = (const float*)d_in[0];
    const float* depth = (const float*)d_in[1];
    const float* Wdown = (const float*)d_in[2];
    const float* Wup   = (const float*)d_in[3];
    const float* W1    = (const float*)d_in[4];
    const float* W2    = (const float*)d_in[5];
    const float* W3    = (const float*)d_in[6];
    const float* W4    = (const float*)d_in[7];
    const float* Wcat  = (const float*)d_in[8];

    const int Bn = 16, N = 1024, C = 256, UP = 768, CAT = 512;

    char* w = (char*)d_ws;
    auto take = [&](size_t bytes) { char* p = w; w += (bytes + 255) & ~(size_t)255; return p; };

    u16* Wdn_b = (u16*)take((size_t)C * UP * 2);
    u16* Wup_b = (u16*)take((size_t)UP * C * 2);
    u16* W1_b  = (u16*)take((size_t)C * C * 2);
    u16* W2_b  = (u16*)take((size_t)C * C * 2);
    u16* W3_b  = (u16*)take((size_t)C * C * 2);
    u16* W4_b  = (u16*)take((size_t)C * C * 2);
    u16* Wc_b  = (u16*)take((size_t)C * CAT * 2);

    float* xy  = (float*)take((size_t)Bn * CAT * N * 4);  // f32 [b][512][N] (residuals)
    u16*   xyT = (u16*)take((size_t)Bn * N * CAT * 2);    // bf16 [b][N][512]
    u16*   QT  = (u16*)take((size_t)Bn * N * C * 2);      // bf16 [b][N][256] = Q^T
    u16*   KT  = (u16*)take((size_t)Bn * N * C * 2);      // bf16 [b][N][256] = K^T
    u16*   Vbf = (u16*)take((size_t)Bn * C * N * 2);      // bf16 [b][256][N] = V
    u16*   refT= (u16*)take((size_t)Bn * N * C * 2);      // bf16 [b][N][256] = refine^T

    // region2: rgbT+depthT (phase 1) aliases Sf[8]+Sbf[8] (attention phase)
    char* region2 = take(2 * (size_t)Bn * UP * N * 2);    // 50.33 MB
    u16*   rgbT = (u16*)region2;
    u16*   depT = rgbT + (size_t)Bn * UP * N;
    float* Sf   = (float*)region2;                        // [8][N][N] f32
    u16*   Sbf  = (u16*)(region2 + (size_t)8 * N * N * 4);// [8][N][N] bf16

    dim3 blk(256);
    auto cvt = [&](const float* in, u16* out, int n) {
        cvt_bf16<<<dim3((n + 255) / 256), blk, 0, stream>>>(in, out, n);
    };
    cvt(Wdown, Wdn_b, C * UP);  cvt(Wup, Wup_b, UP * C);
    cvt(W1, W1_b, C * C);       cvt(W2, W2_b, C * C);
    cvt(W3, W3_b, C * C);       cvt(W4, W4_b, C * C);
    cvt(Wcat, Wc_b, C * CAT);

    transpose_cvt<<<dim3(N / 32, UP / 32, Bn), blk, 0, stream>>>(rgb,   rgbT, UP, N);
    transpose_cvt<<<dim3(N / 32, UP / 32, Bn), blk, 0, stream>>>(depth, depT, UP, N);

    auto gemm = [&](const u16* A, long long lda, long long sAb,
                    const u16* Bm, long long ldb, long long sBb,
                    float* Cf, const float* R, long long ldf, long long sCfb, long long sRb,
                    u16* Cb, long long ldcb, long long sCbb, int tr,
                    int M, int Nn, int K, int bat) {
        dim3 grid(Nn / 128, M / 128, bat);
        gemm_bf16<<<grid, blk, 0, stream>>>(A, lda, sAb, Bm, ldb, sBb,
                                            Cf, R, ldf, sCfb, sRb,
                                            Cb, ldcb, sCbb, tr, K);
    };

    // x = Wdown@rgb, y = Wdown@depth : f32 xy (residual) + bf16 xyT (activations)
    gemm(Wdn_b, UP, 0, rgbT, UP, (long long)N * UP,
         xy, nullptr, N, (long long)CAT * N, 0,
         xyT, CAT, (long long)N * CAT, 1, C, N, UP, Bn);
    gemm(Wdn_b, UP, 0, depT, UP, (long long)N * UP,
         xy + (size_t)C * N, nullptr, N, (long long)CAT * N, 0,
         xyT + C, CAT, (long long)N * CAT, 1, C, N, UP, Bn);

    // Q = Wcat @ [x;y]  -> Q^T bf16
    gemm(Wc_b, CAT, 0, xyT, CAT, (long long)N * CAT,
         nullptr, nullptr, N, 0, 0,
         QT, C, (long long)N * C, 1, C, N, CAT, Bn);

    float* outs[2] = { (float*)d_out, (float*)d_out + (size_t)Bn * UP * N };
    const u16* WKb[2] = { W1_b, W3_b };
    const u16* WVb[2] = { W2_b, W4_b };

    for (int br = 0; br < 2; ++br) {
        const u16*   srcT = xyT + (br ? C : 0);                  // K/V source: x or y
        const float* res  = xy + (br ? 0 : (size_t)C * N);       // residual: y or x

        gemm(WKb[br], C, 0, srcT, CAT, (long long)N * CAT,
             nullptr, nullptr, N, 0, 0,
             KT, C, (long long)N * C, 1, C, N, C, Bn);           // K^T
        gemm(WVb[br], C, 0, srcT, CAT, (long long)N * CAT,
             nullptr, nullptr, N, 0, 0,
             Vbf, N, (long long)C * N, 0, C, N, C, Bn);          // V

        for (int g = 0; g < 2; ++g) {                            // 8 batches per group
            size_t o8 = (size_t)g * 8;
            // S = K^T Q  (A = K^T [N][C], Bt = Q^T [N][C])
            gemm(KT + o8 * N * C, C, (long long)N * C,
                 QT + o8 * N * C, C, (long long)N * C,
                 Sf, nullptr, N, (long long)N * N, 0,
                 nullptr, 0, 0, 0, N, N, C, 8);
            softmax_bf16<<<dim3(N, 8), blk, 0, stream>>>(Sf, Sbf);
            // refine = V P^T + res -> refine^T bf16  (Bt[n][m] = P[n][m] = Sbf row-major)
            gemm(Vbf + o8 * C * N, N, (long long)C * N,
                 Sbf, N, (long long)N * N,
                 nullptr, res + o8 * CAT * N, N, 0, (long long)CAT * N,
                 refT + o8 * N * C, C, (long long)N * C, 1,
                 C, N, N, 8);
        }

        // out = Wup @ refine  (Bt = refine^T [N][C])
        gemm(Wup_b, C, 0, refT, C, (long long)N * C,
             outs[br], nullptr, N, (long long)UP * N, 0,
             nullptr, 0, 0, 0, UP, N, C, Bn);
    }
}
